// LSTMCell_66984309948631
// MI455X (gfx1250) — matrix operand order repaired and checked by reading, not compiled, not run on hardware
//
#include <hip/hip_runtime.h>

typedef __attribute__((ext_vector_type(16))) __bf16 v16bf;
typedef __attribute__((ext_vector_type(8)))  __bf16 bf16x8;
typedef __attribute__((ext_vector_type(4)))  __bf16 bf16x4;
typedef __attribute__((ext_vector_type(8)))  float  v8f;
typedef __attribute__((ext_vector_type(4)))  float  f32x4;
typedef unsigned int __attribute__((ext_vector_type(4))) u32x4;
typedef int          __attribute__((ext_vector_type(8))) i32x8;
typedef int          __attribute__((ext_vector_type(4))) i32x4;

#define B_SZ 16384
#define H_SZ 1024
#define KPH  1024          // K per phase (I or H)
#define MT   128           // M tile per workgroup (8 waves * 16 rows)
#define NT   32            // N tile per workgroup (2 WMMA n-frags)
#define W_ELEMS (4u * 1024u * 1024u)   // elements in one weight tensor [4,1024,1024]

__device__ __forceinline__ float sigmoidf_fast(float x) {
    return 1.0f / (1.0f + __expf(-x));
}
__device__ __forceinline__ float tanhf_fast(float x) {
    float e = __expf(-2.0f * x);
    return (1.0f - e) / (1.0f + e);
}

#define HAVE_TDM (__has_builtin(__builtin_amdgcn_tensor_load_to_lds) && \
                  __has_builtin(__builtin_amdgcn_s_wait_tensorcnt))

// ---------------------------------------------------------------------------
// Pre-pass: convert Wi|Wh (8M f32) -> bf16 workspace (once; weights are
// re-read ~128x by the GEMM, so converting once is strictly cheaper).
// ---------------------------------------------------------------------------
__global__ __launch_bounds__(256)
void cvt_weights(const float* __restrict__ Wi, const float* __restrict__ Wh,
                 __bf16* __restrict__ wbf)
{
    const unsigned gid  = blockIdx.x * 256u + threadIdx.x;
    const unsigned base = gid * 4u;                       // 8M elems / 4 per thread
    const float* src = (base < W_ELEMS) ? (Wi + base) : (Wh + (base - W_ELEMS));
    f32x4 v = *(const f32x4*)src;
    bf16x4 o;
    o[0] = (__bf16)v[0]; o[1] = (__bf16)v[1];
    o[2] = (__bf16)v[2]; o[3] = (__bf16)v[3];
    *(bf16x4*)(wbf + base) = o;
}

#if HAVE_TDM
// ---------------------------------------------------------------------------
// TDM: one 3D tensor_load_to_lds per k-step:
//   dim0 = 32 k (contiguous bf16), dim1 = 32 cols (stride 1024),
//   dim2 = 4 gates (stride 1024*1024). 8KB tile -> LDS ping-pong buffer.
// Descriptor bit layout per CDNA5 ISA §8 (D# groups 0..3).
// Toolchain builtin is the 6-arg form: (g0, g1, g2, g3, i32x8, cpol).
// ---------------------------------------------------------------------------
__device__ __forceinline__ void tdm_load_w_tile(const __bf16* gbase, unsigned ldsOff)
{
    unsigned long long ga = (unsigned long long)(uintptr_t)gbase;
    u32x4 g0;
    g0[0] = 1u;                                            // count=1, user D#
    g0[1] = ldsOff;                                        // lds_addr (bytes)
    g0[2] = (unsigned)(ga & 0xFFFFFFFFu);                  // global_addr[31:0]
    g0[3] = (unsigned)((ga >> 32) & 0x01FFFFFFu)           // global_addr[56:32]
          | (2u << 30);                                    // type=2 ("image")
    i32x8 g1;
    g1[0] = (1 << 16);                                     // data_size=1 (2 bytes)
    g1[1] = (int)(1024u << 16);                            // tensor_dim0[15:0]=1024
    g1[2] = (int)(4096u << 16);                            // dim0[31:16]=0 | tensor_dim1[15:0]=4096
    g1[3] = (int)(32u << 16);                              // dim1[31:16]=0 | tile_dim0=32
    g1[4] = (int)(32u | (4u << 16));                       // tile_dim1=32 | tile_dim2=4
    g1[5] = 1024;                                          // tensor_dim0_stride[31:0]
    g1[6] = 0;                                             // stride0[47:32] | stride1[15:0]
    g1[7] = (int)((1024u * 1024u) >> 16);                  // tensor_dim1_stride[47:16]
    i32x4 g2;
    g2[0] = 16;                                            // tensor_dim2 (>= 4)
    g2[1] = 0;                                             // tensor_dim3 / lds_inc
    g2[2] = 0;                                             // tensor_dim2_stride[31:0]
    g2[3] = 0;                                             // stride2[47:32] | tile_dim3
    i32x4 g3;
    g3[0] = 0; g3[1] = 0; g3[2] = 0; g3[3] = 0;
    i32x8 gx;                                              // extra words (unused)
    gx[0] = 0; gx[1] = 0; gx[2] = 0; gx[3] = 0;
    gx[4] = 0; gx[5] = 0; gx[6] = 0; gx[7] = 0;
    __builtin_amdgcn_tensor_load_to_lds(g0, g1, g2, g3, gx, 0);
}

// ---------------------------------------------------------------------------
// Fast kernel: bf16 weights streamed by TDM (double-buffered), f32
// activations converted inline for the A fragment, f32 WMMA accumulation.
// ---------------------------------------------------------------------------
__global__ __launch_bounds__(256)
void lstm_cell_wmma_tdm(const float* __restrict__ x_in,   // [B, I]
                        const float* __restrict__ h_in,   // [B, H]
                        const float* __restrict__ c_in,   // [B, H]
                        const __bf16* __restrict__ wbf,   // [Wi bf16 | Wh bf16]
                        const float* __restrict__ bi,     // [4, H]
                        float* __restrict__ out)          // [new_h | new_h | new_c]
{
    __shared__ __align__(32) __bf16 ldsW[2 * 4096];       // 2 x 8KB ping-pong

    const int tid  = threadIdx.x;
    const int lane = tid & 31;
    const int wave = tid >> 5;

    const int m_base = blockIdx.x * MT;
    const int n_base = blockIdx.y * NT;
    const int mw     = m_base + wave * 16;

    const int colInLane = lane & 15;
    const int laneHi    = (lane >> 4) & 1;

    v8f acc[4][2];
#pragma unroll
    for (int g = 0; g < 4; ++g)
#pragma unroll
        for (int t = 0; t < 2; ++t)
            acc[g][t] = (v8f){0.f, 0.f, 0.f, 0.f, 0.f, 0.f, 0.f, 0.f};

    const int aRow  = mw + colInLane;
    const int aKoff = laneHi ? 16 : 0;

    // prefetch first weight tile into buffer 0
    if (wave == 0) {
        const __bf16* g0p = wbf + (size_t)n_base * KPH;   // phase 0, k0 = 0
        tdm_load_w_tile(g0p, (unsigned)(uintptr_t)&ldsW[0]);
    }

    for (int ks = 0; ks < 64; ++ks) {                     // 2 phases x 32 k-steps
        const int p     = ks & 1;
        const int phase = ks >> 5;
        const int k0    = (ks & 31) * 32;

        if (wave == 0)
            __builtin_amdgcn_s_wait_tensorcnt(0);         // tile p has landed
        __syncthreads();                                  // all waves done with buf p^1

        if (wave == 0 && ks + 1 < 64) {                   // async prefetch next tile
            const int nph = (ks + 1) >> 5;
            const int nk0 = ((ks + 1) & 31) * 32;
            const __bf16* gp = wbf + (size_t)nph * W_ELEMS
                                   + (size_t)n_base * KPH + nk0;
            tdm_load_w_tile(gp, (unsigned)(uintptr_t)&ldsW[(p ^ 1) * 4096]);
        }

        // ---- A fragment: f32 -> bf16, contiguous k = 16*laneHi + s (B matches) ----
        const float* X = phase ? h_in : x_in;
        const float* ap = X + (size_t)aRow * KPH + k0 + aKoff;
        f32x4 x0 = *(const f32x4*)(ap + 0);
        f32x4 x1 = *(const f32x4*)(ap + 4);
        f32x4 x2 = *(const f32x4*)(ap + 8);
        f32x4 x3 = *(const f32x4*)(ap + 12);
        v16bf av;
#pragma unroll
        for (int j = 0; j < 4; ++j) {
            av[j]      = (__bf16)x0[j];
            av[4 + j]  = (__bf16)x1[j];
            av[8 + j]  = (__bf16)x2[j];
            av[12 + j] = (__bf16)x3[j];
        }

        // ---- preload ALL 8 B fragments (distinct regs so ds_loads pipeline,
        //      decremental s_wait_dscnt instead of wait-to-zero per WMMA) ----
        const int pb = p * 4096;
        v16bf bv[8];
#pragma unroll
        for (int g = 0; g < 4; ++g)
#pragma unroll
            for (int t = 0; t < 2; ++t) {
                const int be = pb + (g * 32 + t * 16 + colInLane) * 32
                             + (laneHi ? 16 : 0);
                bv[g * 2 + t] = *(const v16bf*)&ldsW[be];
            }

        // ---- 4 gates x 2 n-frags of v_wmma_f32_16x16x32_bf16 ----
#pragma unroll
        for (int g = 0; g < 4; ++g)
#pragma unroll
            for (int t = 0; t < 2; ++t)
                acc[g][t] = __builtin_amdgcn_wmma_f32_16x16x32_bf16(
                    false, av, false, bv[g * 2 + t], (short)0, acc[g][t],
                    false, false);
    }

    // ---- epilogue: bias + LSTM elementwise + stores ----
    const size_t BH = (size_t)B_SZ * H_SZ;
#pragma unroll
    for (int t = 0; t < 2; ++t) {
        const int h = n_base + t * 16 + colInLane;
        const float b0 = bi[0 * H_SZ + h];
        const float b1 = bi[1 * H_SZ + h];
        const float b2 = bi[2 * H_SZ + h];
        const float b3 = bi[3 * H_SZ + h];
#pragma unroll
        for (int r = 0; r < 8; ++r) {
            const int m = mw + laneHi * 8 + r;
            const size_t idx = (size_t)m * H_SZ + h;
            const float pi = acc[0][t][r] + b0;
            const float pf = acc[1][t][r] + b1;
            const float pg = acc[2][t][r] + b2;
            const float po = acc[3][t][r] + b3;
            const float iv = sigmoidf_fast(pi);
            const float fv = sigmoidf_fast(pf);
            const float gv = tanhf_fast(pg);
            const float ov = sigmoidf_fast(po);
            const float cn = fv * c_in[idx] + iv * gv;
            const float hn = ov * tanhf_fast(cn);
            out[idx]          = hn;
            out[BH + idx]     = hn;
            out[2 * BH + idx] = cn;
        }
    }
}
#endif // HAVE_TDM

// ---------------------------------------------------------------------------
// Fallback: self-contained kernel (round-1 style, no workspace needed).
// ---------------------------------------------------------------------------
__global__ __launch_bounds__(256)
void lstm_cell_wmma(const float* __restrict__ x_in,
                    const float* __restrict__ h_in,
                    const float* __restrict__ c_in,
                    const float* __restrict__ Wi,
                    const float* __restrict__ bi,
                    const float* __restrict__ Wh,
                    float* __restrict__ out)
{
    __shared__ __align__(32) __bf16 ldsW[4 * 32 * 32];

    const int tid  = threadIdx.x;
    const int lane = tid & 31;
    const int wave = tid >> 5;

    const int m_base = blockIdx.x * MT;
    const int n_base = blockIdx.y * NT;
    const int mw     = m_base + wave * 16;

    const int colInLane = lane & 15;
    const int laneHi    = (lane >> 4) & 1;

    v8f acc[4][2];
#pragma unroll
    for (int g = 0; g < 4; ++g)
#pragma unroll
        for (int t = 0; t < 2; ++t)
            acc[g][t] = (v8f){0.f, 0.f, 0.f, 0.f, 0.f, 0.f, 0.f, 0.f};

    const int sg  = tid >> 6;
    const int scl = (tid >> 1) & 31;
    const int skh = tid & 1;

    const int aRow  = mw + colInLane;
    const int aKoff = laneHi ? 16 : 0;

    for (int phase = 0; phase < 2; ++phase) {
        const float* __restrict__ X = phase ? h_in : x_in;
        const float* __restrict__ W = phase ? Wh   : Wi;

        const float* aBase = X + (size_t)aRow * KPH + aKoff;
        const float* sBase = W + ((size_t)(sg * H_SZ + n_base + scl)) * KPH + skh * 16;
        __bf16* sDst = &ldsW[(sg * 32 + scl) * 32 + skh * 16];

        for (int k0 = 0; k0 < KPH; k0 += 32) {
            {
                const float* src = sBase + k0;
                f32x4 w0 = *(const f32x4*)(src + 0);
                f32x4 w1 = *(const f32x4*)(src + 4);
                f32x4 w2 = *(const f32x4*)(src + 8);
                f32x4 w3 = *(const f32x4*)(src + 12);
                bf16x8 p0, p1;
#pragma unroll
                for (int j = 0; j < 4; ++j) {
                    p0[j]     = (__bf16)w0[j];
                    p0[4 + j] = (__bf16)w1[j];
                    p1[j]     = (__bf16)w2[j];
                    p1[4 + j] = (__bf16)w3[j];
                }
                *(bf16x8*)(sDst)     = p0;
                *(bf16x8*)(sDst + 8) = p1;
            }
            __syncthreads();

            v16bf av;
            {
                const float* ap = aBase + k0;
                f32x4 x0 = *(const f32x4*)(ap + 0);
                f32x4 x1 = *(const f32x4*)(ap + 4);
                f32x4 x2 = *(const f32x4*)(ap + 8);
                f32x4 x3 = *(const f32x4*)(ap + 12);
#pragma unroll
                for (int j = 0; j < 4; ++j) {
                    av[j]      = (__bf16)x0[j];
                    av[4 + j]  = (__bf16)x1[j];
                    av[8 + j]  = (__bf16)x2[j];
                    av[12 + j] = (__bf16)x3[j];
                }
            }

#pragma unroll
            for (int g = 0; g < 4; ++g) {
#pragma unroll
                for (int t = 0; t < 2; ++t) {
                    const int be = (g * 32 + t * 16 + colInLane) * 32 + (laneHi ? 16 : 0);
                    v16bf bv = *(const v16bf*)&ldsW[be];
                    acc[g][t] = __builtin_amdgcn_wmma_f32_16x16x32_bf16(
                        false, av, false, bv, (short)0, acc[g][t], false, false);
                }
            }
            __syncthreads();
        }
    }

    const size_t BH = (size_t)B_SZ * H_SZ;
#pragma unroll
    for (int t = 0; t < 2; ++t) {
        const int h = n_base + t * 16 + colInLane;
        const float b0 = bi[0 * H_SZ + h];
        const float b1 = bi[1 * H_SZ + h];
        const float b2 = bi[2 * H_SZ + h];
        const float b3 = bi[3 * H_SZ + h];
#pragma unroll
        for (int r = 0; r < 8; ++r) {
            const int m = mw + laneHi * 8 + r;
            const size_t idx = (size_t)m * H_SZ + h;
            const float pi = acc[0][t][r] + b0;
            const float pf = acc[1][t][r] + b1;
            const float pg = acc[2][t][r] + b2;
            const float po = acc[3][t][r] + b3;
            const float iv = sigmoidf_fast(pi);
            const float fv = sigmoidf_fast(pf);
            const float gv = tanhf_fast(pg);
            const float ov = sigmoidf_fast(po);
            const float cn = fv * c_in[idx] + iv * gv;
            const float hn = ov * tanhf_fast(cn);
            out[idx]          = hn;
            out[BH + idx]     = hn;
            out[2 * BH + idx] = cn;
        }
    }
}

extern "C" void kernel_launch(void* const* d_in, const int* in_sizes, int n_in,
                              void* d_out, int out_size, void* d_ws, size_t ws_size,
                              hipStream_t stream) {
    (void)in_sizes; (void)n_in; (void)out_size;
    const float* x_in = (const float*)d_in[0];
    const float* h_in = (const float*)d_in[1];
    const float* c_in = (const float*)d_in[2];
    const float* Wi   = (const float*)d_in[3];
    const float* bi   = (const float*)d_in[4];
    const float* Wh   = (const float*)d_in[5];
    float* out = (float*)d_out;

    dim3 grid(B_SZ / MT, H_SZ / NT);   // 128 x 32
    dim3 block(256, 1, 1);

#if HAVE_TDM
    if (ws_size >= (size_t)2 * W_ELEMS * sizeof(__bf16)) {   // 16 MB
        __bf16* wbf = (__bf16*)d_ws;
        // 8M weight elems, 4 per thread, 256 threads/block -> 8192 blocks
        cvt_weights<<<dim3(8192), block, 0, stream>>>(Wi, Wh, wbf);
        lstm_cell_wmma_tdm<<<grid, block, 0, stream>>>(x_in, h_in, c_in, wbf, bi, out);
        return;
    }
#endif
    (void)d_ws; (void)ws_size;
    lstm_cell_wmma<<<grid, block, 0, stream>>>(x_in, h_in, c_in, Wi, bi, Wh, out);
}
